// AnyAttention_21122649162368
// MI455X (gfx1250) — matrix operand order repaired and checked by reading, not compiled, not run on hardware
//
#include <hip/hip_runtime.h>

// ---------------------------------------------------------------------------
// Sparse top-k attention for MI455X (gfx1250, wave32, WMMA).
//
// Pipeline:
//   1) ln_f16_kernel        : LayerNorm(q|k|v) -> f16 staging
//   2) f32_to_f16_kernel    : weight conversion wq/wk/wv/wp -> f16
//   3) gemm16x128<true,768> : qh/kh/vh = LN(x) @ W^T + b   (f16 out, f32 accum WMMA)
//   4) gemm16x128<false,96> : raw logits (no scale) written DIRECTLY into the
//                             attn region of d_out (saves a 268MB scratch pass)
//   5) topk_ctx_kernel      : per row: top-10 via iterated wave argmax, softmax/tau,
//                             rewrite row in-place as zeros + 10 values, and fuse
//                             the sparse attn@V gather -> ctx (f16)
//   6) gemm16x128<false,768>: out = ctx @ wp^T + bp  (fp32 into d_out)
//
// GEMM: 16x128 wave tile (8 accumulators, 8 independent v_wmma per A-load).
// amdgpu_waves_per_eu(1,2) relaxes the occupancy-driven VGPR cap (the default
// heuristic pinned the kernel at 64 VGPRs, forcing loadcnt-0 drains before
// every WMMA); 2 waves/SIMD is plenty since operands are L2-resident.
// ---------------------------------------------------------------------------

#define DIMC   768
#define BATCH  8
#define NSEQ   1024
#define HEADS  8
#define HDIM   96
#define NTOK   (BATCH * NSEQ)          // 8192
#define TOPK   10
#define SCALE_F 0.10206207261596577f   // 96^-0.5
#define OUT_ELEMS ((size_t)NTOK * DIMC)           // 6291456

typedef _Float16 v8h  __attribute__((ext_vector_type(8)));
typedef _Float16 v16h __attribute__((ext_vector_type(16)));
typedef float    v8f  __attribute__((ext_vector_type(8)));

static __device__ inline v16h cat16(v8h lo, v8h hi) {
    return __builtin_shufflevector(lo, hi, 0,1,2,3,4,5,6,7,8,9,10,11,12,13,14,15);
}

// ---------------------------------------------------------------------------
// LayerNorm (row of 768) -> f16
// ---------------------------------------------------------------------------
__global__ __launch_bounds__(256)
void ln_f16_kernel(const float* __restrict__ x, const float* __restrict__ g,
                   const float* __restrict__ bta, _Float16* __restrict__ y)
{
    const int row = blockIdx.x;
    const int t = threadIdx.x;
    const float* xr = x + (size_t)row * DIMC;
    float v0 = xr[t], v1 = xr[t + 256], v2 = xr[t + 512];
    __shared__ float sh1[256], sh2[256];
    sh1[t] = v0 + v1 + v2;
    sh2[t] = v0 * v0 + v1 * v1 + v2 * v2;
    __syncthreads();
    for (int o = 128; o > 0; o >>= 1) {
        if (t < o) { sh1[t] += sh1[t + o]; sh2[t] += sh2[t + o]; }
        __syncthreads();
    }
    const float mean = sh1[0] * (1.0f / DIMC);
    const float var  = sh2[0] * (1.0f / DIMC) - mean * mean;
    const float rs   = rsqrtf(var + 1e-5f);
    _Float16* yr = y + (size_t)row * DIMC;
    yr[t]       = (_Float16)((v0 - mean) * rs * g[t]       + bta[t]);
    yr[t + 256] = (_Float16)((v1 - mean) * rs * g[t + 256] + bta[t + 256]);
    yr[t + 512] = (_Float16)((v2 - mean) * rs * g[t + 512] + bta[t + 512]);
}

__global__ __launch_bounds__(256)
void f32_to_f16_kernel(const float* __restrict__ s, _Float16* __restrict__ d, int n)
{
    int i = blockIdx.x * 256 + threadIdx.x;
    if (i < n) d[i] = (_Float16)s[i];
}

// ---------------------------------------------------------------------------
// WMMA GEMM: C = A(MxK, row-major) * B(NxK, row-major, i.e. B^T) + bias
// Each wave computes a 16x128 tile: 8x v_wmma_f32_16x16x32_f16 per K-step.
// A-fragment (ISA 7.12.2): lane<16 -> row M=lane, K chunks {k..k+7, k+16..k+23};
//                          lane>=16 -> same row set, K chunks {k+8.., k+24..}.
// B-fragment: lane's column n = lane&15 (+16*j subtile), same K chunks as A:
//             {k+8*(lane>>4).., k+16+8*(lane>>4)..}.
// Grid: x = wave-tiles/8, y/z = batch dims with independent strides.
// ---------------------------------------------------------------------------
template<bool OUT_HALF, int KT>
__global__ __launch_bounds__(256)
__attribute__((amdgpu_waves_per_eu(1, 2)))
void gemm16x128(const _Float16* __restrict__ A, const _Float16* __restrict__ B,
                void* __restrict__ Cout, const float* __restrict__ bias,
                int M, int N, int lda, int ldb, int ldc,
                size_t sAy, size_t sAz, size_t sBy, size_t sBz,
                size_t sCy, size_t sCz)
{
    const int lane = threadIdx.x & 31;
    const int wave = threadIdx.x >> 5;
    const int tilesN = N >> 7;          // N/128
    const int tilesM = M >> 4;          // M/16
    const int tile = blockIdx.x * 8 + wave;
    if (tile >= tilesM * tilesN) return;          // wave-uniform exit, EXEC stays full
    const int tm = tile / tilesN;
    const int tn = tile % tilesN;

    A += blockIdx.y * sAy + blockIdx.z * sAz;
    B += blockIdx.y * sBy + blockIdx.z * sBz;

    const int half = lane >> 4;   // 0 or 1
    const int l15  = lane & 15;
    const _Float16* aRow = A + (size_t)(tm * 16 + l15) * lda + half * 8;
    const _Float16* bRow = B + (size_t)(tn * 128 + l15) * ldb + half * 8;

    // pull the tile's cachelines toward the WGP before the load burst
    __builtin_prefetch(aRow, 0, 3);
    #pragma unroll
    for (int j = 0; j < 8; ++j)
        __builtin_prefetch(bRow + (size_t)j * 16 * ldb, 0, 3);

    v8f acc[8] = {v8f{}, v8f{}, v8f{}, v8f{}, v8f{}, v8f{}, v8f{}, v8f{}};
    #pragma unroll
    for (int k = 0; k < KT; k += 32) {
        // loads grouped first -> load clause per K-step, then 8 independent WMMAs
        const v16h a = cat16(*(const v8h*)(aRow + k), *(const v8h*)(aRow + k + 16));
        v16h b[8];
        #pragma unroll
        for (int j = 0; j < 8; ++j) {
            const _Float16* bp = bRow + (size_t)j * 16 * ldb + k;
            b[j] = cat16(*(const v8h*)(bp), *(const v8h*)(bp + 16));
        }
        #pragma unroll
        for (int j = 0; j < 8; ++j)
            acc[j] = __builtin_amdgcn_wmma_f32_16x16x32_f16(
                false, a, false, b[j], (short)0, acc[j], false, false);
    }

    float*    cf = (float*)Cout;
    _Float16* ch = (_Float16*)Cout;
    const size_t cbase = blockIdx.y * sCy + blockIdx.z * sCz;
    #pragma unroll
    for (int j = 0; j < 8; ++j) {
        const int n = tn * 128 + j * 16 + l15;
        const float bb = bias ? bias[n] : 0.0f;
        #pragma unroll
        for (int r = 0; r < 8; ++r) {
            const int m = tm * 16 + half * 8 + r;   // C/D layout: M = r + 8*(lane>=16)
            const float v = acc[j][r] + bb;
            const size_t idx = cbase + (size_t)m * ldc + n;
            if (OUT_HALF) ch[idx] = (_Float16)v;
            else          cf[idx] = v;
        }
    }
}

// ---------------------------------------------------------------------------
// Top-10 select + softmax(tau) + in-place sparse rewrite + fused attn@V.
// One wave32 per logits row (1024 values, read as 8x B128 per lane).
// Position mapping: vals[4*j4+c] holds logits index p = j4*128 + lane*4 + c.
// ---------------------------------------------------------------------------
__global__ __launch_bounds__(256)
void topk_ctx_kernel(float* __restrict__ attn, const _Float16* __restrict__ vh,
                     _Float16* __restrict__ ctx, const float* __restrict__ taup)
{
    const int lane = threadIdx.x & 31;
    const int wave = threadIdx.x >> 5;
    const int row  = blockIdx.x * 8 + wave;       // 0 .. 65535 == (b,h,q)
    const int q  = row & (NSEQ - 1);
    const int bh = row >> 10;
    const int h  = bh & (HEADS - 1);
    const int b  = bh >> 3;

    float* rp = attn + (size_t)row * NSEQ;
    float4* r4 = (float4*)rp;

    float vals[32];
    #pragma unroll
    for (int j4 = 0; j4 < 8; ++j4) {
        const float4 t = r4[j4 * 32 + lane];
        vals[4 * j4 + 0] = t.x * SCALE_F;
        vals[4 * j4 + 1] = t.y * SCALE_F;
        vals[4 * j4 + 2] = t.z * SCALE_F;
        vals[4 * j4 + 3] = t.w * SCALE_F;
    }

    float topv[TOPK];
    int   topi[TOPK];
    #pragma unroll
    for (int it = 0; it < TOPK; ++it) {
        float bv = -3.4e38f;
        int   bi = 1 << 30;
        #pragma unroll
        for (int j = 0; j < 32; ++j) {
            const int p = (j >> 2) * 128 + lane * 4 + (j & 3);
            if (vals[j] > bv || (vals[j] == bv && p < bi)) { bv = vals[j]; bi = p; }
        }
        for (int o = 16; o > 0; o >>= 1) {        // wave32 argmax reduce
            const float ov = __shfl_xor(bv, o, 32);
            const int   oi = __shfl_xor(bi, o, 32);
            if (ov > bv || (ov == bv && oi < bi)) { bv = ov; bi = oi; }
        }
        topv[it] = bv;
        topi[it] = bi;
        // knock out the winner: owner lane = (bi>>2)&31, reg idx = (bi>>7)*4 + (bi&3)
        const int ll = (bi >> 2) & 31;
        const int jj = ((bi >> 7) << 2) | (bi & 3);
        #pragma unroll
        for (int j = 0; j < 32; ++j)
            if (j == jj && lane == ll) vals[j] = -3.4e38f;
    }

    const float tau = fmaxf(taup[0], 1e-8f);
    const float inv = 1.0f / tau;
    float e[TOPK], ssum = 0.0f;
    #pragma unroll
    for (int i = 0; i < TOPK; ++i) { e[i] = __expf((topv[i] - topv[0]) * inv); ssum += e[i]; }
    const float rs = 1.0f / ssum;

    // zero the whole row with B128 stores, then scatter the 10 softmax values
    const float4 z = make_float4(0.f, 0.f, 0.f, 0.f);
    #pragma unroll
    for (int j = 0; j < 8; ++j) r4[j * 32 + lane] = z;
    asm volatile("s_wait_storecnt 0x0" ::: "memory");   // zeros land before scatter

    float wl = 0.0f;
    int   ti = 0;
    #pragma unroll
    for (int i = 0; i < TOPK; ++i)
        if (lane == i) { wl = e[i] * rs; ti = topi[i]; }
    if (lane < TOPK) rp[ti] = wl;

    // fused sparse attn @ V : ctx[b,q,h*96+col] = sum_i w_i * vh[b,topi_i,h*96+col]
    const _Float16* vb = vh  + (size_t)b * NSEQ * DIMC + h * HDIM;
    _Float16*       cb = ctx + (size_t)(b * NSEQ + q) * DIMC + h * HDIM;
    #pragma unroll
    for (int cc = 0; cc < 3; ++cc) {
        const int col = cc * 32 + lane;
        float acc = 0.0f;
        #pragma unroll
        for (int i = 0; i < TOPK; ++i)
            acc += (e[i] * rs) * (float)vb[(size_t)topi[i] * DIMC + col];
        cb[col] = (_Float16)acc;
    }
}

// ---------------------------------------------------------------------------
extern "C" void kernel_launch(void* const* d_in, const int* in_sizes, int n_in,
                              void* d_out, int out_size, void* d_ws, size_t ws_size,
                              hipStream_t stream)
{
    const float* q     = (const float*)d_in[0];
    const float* k     = (const float*)d_in[1];
    const float* v     = (const float*)d_in[2];
    const float* lnq_g = (const float*)d_in[3];
    const float* lnq_b = (const float*)d_in[4];
    const float* lnk_g = (const float*)d_in[5];
    const float* lnk_b = (const float*)d_in[6];
    const float* lnv_g = (const float*)d_in[7];
    const float* lnv_b = (const float*)d_in[8];
    const float* wq    = (const float*)d_in[9];
    const float* bq    = (const float*)d_in[10];
    const float* wk    = (const float*)d_in[11];
    const float* bk    = (const float*)d_in[12];
    const float* wv    = (const float*)d_in[13];
    const float* bv    = (const float*)d_in[14];
    const float* wp    = (const float*)d_in[15];
    const float* bp    = (const float*)d_in[16];
    const float* tau   = (const float*)d_in[17];
    (void)in_sizes; (void)n_in; (void)out_size; (void)ws_size;

    // workspace carve (~93 MB)
    char* w = (char*)d_ws;
    auto alloc = [&](size_t bytes) { void* p = (void*)w; w += (bytes + 255) & ~(size_t)255; return p; };
    const size_t plane = (size_t)NTOK * DIMC * sizeof(_Float16);   // 12.6 MB
    const size_t wsz   = (size_t)DIMC * DIMC * sizeof(_Float16);   // 1.18 MB
    _Float16* qn   = (_Float16*)alloc(plane);
    _Float16* kn   = (_Float16*)alloc(plane);
    _Float16* vn   = (_Float16*)alloc(plane);
    _Float16* qh   = (_Float16*)alloc(plane);
    _Float16* kh   = (_Float16*)alloc(plane);
    _Float16* vh   = (_Float16*)alloc(plane);
    _Float16* ctx  = (_Float16*)alloc(plane);
    _Float16* wq16 = (_Float16*)alloc(wsz);
    _Float16* wk16 = (_Float16*)alloc(wsz);
    _Float16* wv16 = (_Float16*)alloc(wsz);
    _Float16* wp16 = (_Float16*)alloc(wsz);

    float* out_p  = (float*)d_out;
    float* attn_p = (float*)d_out + OUT_ELEMS;

    // 1) LayerNorm -> f16
    ln_f16_kernel<<<NTOK, 256, 0, stream>>>(q, lnq_g, lnq_b, qn);
    ln_f16_kernel<<<NTOK, 256, 0, stream>>>(k, lnk_g, lnk_b, kn);
    ln_f16_kernel<<<NTOK, 256, 0, stream>>>(v, lnv_g, lnv_b, vn);

    // 2) weights -> f16
    const int WN = DIMC * DIMC;
    f32_to_f16_kernel<<<(WN + 255) / 256, 256, 0, stream>>>(wq, wq16, WN);
    f32_to_f16_kernel<<<(WN + 255) / 256, 256, 0, stream>>>(wk, wk16, WN);
    f32_to_f16_kernel<<<(WN + 255) / 256, 256, 0, stream>>>(wv, wv16, WN);
    f32_to_f16_kernel<<<(WN + 255) / 256, 256, 0, stream>>>(wp, wp16, WN);

    // 3) projections: (8192x768)*(768x768)^T ; (8192/16)*(768/128)=3072 tiles -> 384 blocks
    const int projBlocks = ((NTOK / 16) * (DIMC / 128) + 7) / 8;
    dim3 gP(projBlocks, 1, 1);
    gemm16x128<true, DIMC><<<gP, 256, 0, stream>>>(qn, wq16, qh, bq,
        NTOK, DIMC, DIMC, DIMC, DIMC, 0, 0, 0, 0, 0, 0);
    gemm16x128<true, DIMC><<<gP, 256, 0, stream>>>(kn, wk16, kh, bk,
        NTOK, DIMC, DIMC, DIMC, DIMC, 0, 0, 0, 0, 0, 0);
    gemm16x128<true, DIMC><<<gP, 256, 0, stream>>>(vn, wv16, vh, bv,
        NTOK, DIMC, DIMC, DIMC, DIMC, 0, 0, 0, 0, 0, 0);

    // 4) logits (unscaled) straight into attn region: 64 batches of (1024x96)*(96x1024)
    //    (1024/16)*(1024/128)=512 tiles -> 64 blocks; grid y = head, z = batch
    const int logitBlocks = ((NSEQ / 16) * (NSEQ / 128) + 7) / 8;
    dim3 gL(logitBlocks, HEADS, BATCH);
    gemm16x128<false, HDIM><<<gL, 256, 0, stream>>>(qh, kh, attn_p, nullptr,
        NSEQ, NSEQ, DIMC, DIMC, NSEQ,
        /*sAy=h*/ (size_t)HDIM, /*sAz=b*/ (size_t)NSEQ * DIMC,
        /*sBy=h*/ (size_t)HDIM, /*sBz=b*/ (size_t)NSEQ * DIMC,
        /*sCy=h*/ (size_t)NSEQ * NSEQ, /*sCz=b*/ (size_t)HEADS * NSEQ * NSEQ);

    // 5) top-10 + softmax + in-place sparse rewrite + fused attn@V -> ctx (f16)
    topk_ctx_kernel<<<(BATCH * HEADS * NSEQ) / 8, 256, 0, stream>>>(attn_p, vh, ctx, tau);

    // 6) output projection -> fp32 out
    gemm16x128<false, DIMC><<<gP, 256, 0, stream>>>(ctx, wp16, out_p, bp,
        NTOK, DIMC, DIMC, DIMC, DIMC, 0, 0, 0, 0, 0, 0);
}
